// ThreeStageFFN_82068235092427
// MI455X (gfx1250) — matrix operand order repaired and checked by reading, not compiled, not run on hardware
//
#include <hip/hip_runtime.h>
#include <hip/hip_bf16.h>
#include <math.h>

// ---------------------------------------------------------------------------
// ThreeStageFFN for MI455X (gfx1250, wave32, WMMA).
// dims (from reference): d_model=512, n_in=2048, n_proc=1024, n_out=2048,
// d_pv=256, tokens = B*S = 2048, K_INPUT=256, K_PROCESS=128, K_OUTPUT=256.
//
// Strategy: every stage is a dense [16,K] x W^T[N,K] GEMM using
// v_wmma_f32_16x16x32_bf16, plus an exact per-row radix-select (4x8-bit
// passes on order-isomorphic uint keys) that replaces topk+scatter with
// threshold masking. Softmax (stage 2) is computed over the selected set.
// One workgroup (8 waves) per 16-token tile; 128 workgroups.
// ---------------------------------------------------------------------------

typedef __attribute__((ext_vector_type(16))) __bf16 v16bf;
typedef __attribute__((ext_vector_type(8)))  float  v8f;

#define WAVES 8
#define BLOCK 256

// ---- model dims (hard-coded to the reference) ----
#define D_MODEL 512
#define N_IN    2048
#define N_PROC  1024
#define D_PV    256
#define N_OUT   2048
#define TOKENS  2048
#define K_IN    256
#define K_PR    128
#define K_OUT   256

// ---- bf16 workspace layout (element offsets) ----
#define WS_IP   0                                   // [N_IN ][D_MODEL] direct
#define WS_PW   (WS_IP  + N_IN * D_MODEL)           // [N_PROC][N_IN]   direct
#define WS_PVT  (WS_PW  + N_PROC * N_IN)            // [D_PV ][N_PROC]  transposed
#define WS_OW   (WS_PVT + D_PV * N_PROC)            // [N_OUT][D_PV]    direct
#define WS_OPT  (WS_OW  + N_OUT * D_PV)             // [D_MODEL][N_OUT] transposed
#define WS_END  (WS_OPT + D_MODEL * N_OUT)          // 4,980,736 elems (~9.5 MB)

// ---- dynamic LDS layout (bytes) ----
#define SM_ACTSF 0                                  // float [16][2048]  128 KB
#define SM_ACTSB (SM_ACTSF + 16*2048*4)             // bf16  [16][2048]   64 KB
#define SM_XB    (SM_ACTSB + 16*2048*2)             // bf16  [16][512]    16 KB
#define SM_GF    (SM_XB    + 16*512*2)              // float [16][256]    16 KB
#define SM_GB    (SM_GF    + 16*256*4)              // bf16  [16][256]     8 KB
#define SM_HIST  (SM_GB    + 16*256*2)              // uint  [16][256]    16 KB
#define SM_RSEL  (SM_HIST  + 16*256*4)              // uint  [16][2]
#define SM_BYTES (SM_RSEL  + 16*2*4)                // ~248 KB < 320 KB/WGP

__device__ __forceinline__ float gelu_exact(float v) {
  return 0.5f * v * (1.0f + erff(v * 0.70710678118654752f));
}

// order-isomorphic mapping: f < g  <=>  ordKey(f) < ordKey(g)
__device__ __forceinline__ unsigned ordKey(float f) {
  unsigned u = __float_as_uint(f);
  return (u & 0x80000000u) ? ~u : (u | 0x80000000u);
}

union FragU { uint4 u[2]; v16bf v; };

// A fragment (16x32 bf16): lane L, row=L&15, half=L>>4,
// elems e -> K = half*16 + e.
// p must point at  base + row*K + k + half*16
__device__ __forceinline__ v16bf loadFragA(const __hip_bfloat16* p) {
  FragU f;
  f.u[0] = *(const uint4*)(p);
  f.u[1] = *(const uint4*)(p + 8);
  return f.v;
}

// B fragment (32x16 bf16): lane L, n=L&15, elems e -> K = (L>>4)*16 + e.
// p must point at  W + (ncol)*K + k + half*16  -> 32 contiguous bytes.
__device__ __forceinline__ v16bf loadFragB(const __hip_bfloat16* p) {
  FragU f;
  f.u[0] = *(const uint4*)(p);
  f.u[1] = *(const uint4*)(p + 8);
  return f.v;
}

// D = [16,K]x[K,N] with B[k][n] = W[n][k] (W is [N][K] bf16, global).
// A is [16][K] bf16 (LDS). D is float, row stride N (LDS or global).
// Waves split N into 16-wide tiles, two accumulators in flight per wave.
template<bool GELU>
__device__ __forceinline__ void wave_gemm(const __hip_bfloat16* __restrict__ W,
                                          const __hip_bfloat16* A,
                                          int K, int N, float* D) {
  const int wave = threadIdx.x >> 5;
  const int lane = threadIdx.x & 31;
  const int row  = lane & 15;
  const int half = lane >> 4;
  const int numTiles = N >> 4;
  const __hip_bfloat16* arow = A + row * K + half * 16;
  for (int nt = wave; nt < numTiles; nt += 2 * WAVES) {
    const int nt1 = nt + WAVES;
    v8f acc0 = {}; v8f acc1 = {};
    const __hip_bfloat16* w0 = W + (size_t)(nt  * 16 + row) * K + half * 16;
    const __hip_bfloat16* w1 = W + (size_t)(nt1 * 16 + row) * K + half * 16;
    for (int k = 0; k < K; k += 32) {
      v16bf a  = loadFragA(arow + k);
      v16bf b0 = loadFragB(w0 + k);
      v16bf b1 = loadFragB(w1 + k);
      acc0 = __builtin_amdgcn_wmma_f32_16x16x32_bf16(false, a, false, b0,
                                                     (short)0, acc0, false, false);
      acc1 = __builtin_amdgcn_wmma_f32_16x16x32_bf16(false, a, false, b1,
                                                     (short)0, acc1, false, false);
    }
    // D layout: VGPR r, lane L -> M = half*8 + r, Ncol = lane&15
#pragma unroll
    for (int r = 0; r < 8; ++r) {
      float v0 = acc0[r], v1 = acc1[r];
      if (GELU) { v0 = gelu_exact(v0); v1 = gelu_exact(v1); }
      D[(size_t)(half * 8 + r) * N + nt  * 16 + row] = v0;
      D[(size_t)(half * 8 + r) * N + nt1 * 16 + row] = v1;
    }
  }
}

// Per-row exact k-th-largest selection (radix select, 4x8-bit passes) then
// masked write to bf16. mode 0: keep raw values; mode 1: softmax weights
// over the selected set. One wave owns rows (wave) and (wave+8).
__device__ __forceinline__ void select_mask(const float* actsF,
                                            __hip_bfloat16* actsB,
                                            int n, int k,
                                            unsigned* hist, unsigned* rowsel,
                                            int mode) {
  const int wave = threadIdx.x >> 5;
  const int lane = threadIdx.x & 31;
#pragma unroll
  for (int rr = 0; rr < 2; ++rr) {
    const int row = wave + rr * 8;
    unsigned* h = hist + row * 256;
    const float* af = actsF + (size_t)row * n;
    unsigned prefix = 0;
    int kneed = k;
    for (int pass = 0; pass < 4; ++pass) {
      const int shift = 24 - 8 * pass;
      for (int b = lane; b < 256; b += 32) h[b] = 0;
      __builtin_amdgcn_wave_barrier();
      for (int i = lane; i < n; i += 32) {
        unsigned u = ordKey(af[i]);
        bool match = (pass == 0) || ((u >> (shift + 8)) == (prefix >> (shift + 8)));
        if (match) atomicAdd(&h[(u >> shift) & 255u], 1u);
      }
      __builtin_amdgcn_wave_barrier();
      if (lane == 0) {
        unsigned cum = 0; int b = 255;
        for (; b > 0; --b) {
          unsigned c = h[b];
          if (cum + c >= (unsigned)kneed) break;
          cum += c;
        }
        rowsel[row * 2]     = (unsigned)b;
        rowsel[row * 2 + 1] = cum;
      }
      __builtin_amdgcn_wave_barrier();
      unsigned b = rowsel[row * 2];
      kneed -= (int)rowsel[row * 2 + 1];
      prefix |= b << shift;
      __builtin_amdgcn_wave_barrier();
    }
    const unsigned thresh = prefix;  // exact key of the k-th largest
    __hip_bfloat16* ab = actsB + (size_t)row * n;
    if (mode == 0) {
      for (int i = lane; i < n; i += 32) {
        float f = af[i];
        ab[i] = __float2bfloat16(ordKey(f) >= thresh ? f : 0.0f);
      }
    } else {
      float m = -1e30f;
      for (int i = lane; i < n; i += 32) {
        float f = af[i];
        if (ordKey(f) >= thresh) m = fmaxf(m, f);
      }
#pragma unroll
      for (int off = 16; off; off >>= 1) m = fmaxf(m, __shfl_xor(m, off, 32));
      float s = 0.0f;
      for (int i = lane; i < n; i += 32) {
        float f = af[i];
        if (ordKey(f) >= thresh) s += __expf(f - m);
      }
#pragma unroll
      for (int off = 16; off; off >>= 1) s += __shfl_xor(s, off, 32);
      const float inv = 1.0f / s;
      for (int i = lane; i < n; i += 32) {
        float f = af[i];
        bool keep = ordKey(f) >= thresh;
        ab[i] = __float2bfloat16(keep ? __expf(f - m) * inv : 0.0f);
      }
    }
  }
}

// ---- weight preprocessing: f32 -> bf16 (direct / transposed) ----
__global__ void ffn3_cvt_direct(const float* __restrict__ src,
                                __hip_bfloat16* __restrict__ dst, int n) {
  int i = blockIdx.x * blockDim.x + threadIdx.x;
  if (i < n) dst[i] = __float2bfloat16(src[i]);
}

__global__ void ffn3_cvt_transpose(const float* __restrict__ src,
                                   __hip_bfloat16* __restrict__ dst,
                                   int R, int C) {  // src[R][C] -> dst[C][R]
  int i = blockIdx.x * blockDim.x + threadIdx.x;
  if (i < R * C) {
    int r = i / C, c = i - r * C;
    dst[(size_t)c * R + r] = __float2bfloat16(src[i]);
  }
}

// ---- main fused kernel: one workgroup per 16-token tile ----
__global__ void __launch_bounds__(BLOCK)
ffn3_main_kernel(const float* __restrict__ x,
                 const __hip_bfloat16* __restrict__ ws,
                 float* __restrict__ out) {
  extern __shared__ char smem[];
  float*          actsF  = (float*)(smem + SM_ACTSF);
  __hip_bfloat16* actsB  = (__hip_bfloat16*)(smem + SM_ACTSB);
  __hip_bfloat16* xb     = (__hip_bfloat16*)(smem + SM_XB);
  float*          gF     = (float*)(smem + SM_GF);
  __hip_bfloat16* gB     = (__hip_bfloat16*)(smem + SM_GB);
  unsigned*       hist   = (unsigned*)(smem + SM_HIST);
  unsigned*       rowsel = (unsigned*)(smem + SM_RSEL);

  const __hip_bfloat16* ip  = ws + WS_IP;
  const __hip_bfloat16* pw  = ws + WS_PW;
  const __hip_bfloat16* pvt = ws + WS_PVT;
  const __hip_bfloat16* ow  = ws + WS_OW;
  const __hip_bfloat16* opt = ws + WS_OPT;

  const int tokenBase = blockIdx.x * 16;

  // load + convert x tile [16][512] -> bf16 LDS
  for (int i = threadIdx.x; i < 16 * D_MODEL; i += BLOCK)
    xb[i] = __float2bfloat16(x[(size_t)tokenBase * D_MODEL + i]);
  __syncthreads();

  // Stage 1: A1 = gelu(X @ IP^T)  [16,2048]
  wave_gemm<true>(ip, xb, D_MODEL, N_IN, actsF);
  __syncthreads();
  // top-256 threshold mask -> R (bf16)
  select_mask(actsF, actsB, N_IN, K_IN, hist, rowsel, 0);
  __syncthreads();

  // Stage 2: A2 = gelu(R @ PW^T)  [16,1024]
  wave_gemm<true>(pw, actsB, N_IN, N_PROC, actsF);
  __syncthreads();
  // top-128 + softmax weights -> W2 (bf16)
  select_mask(actsF, actsB, N_PROC, K_PR, hist, rowsel, 1);
  __syncthreads();

  // Stage 2b: G = W2 @ PV  (PVT is [256][1024] so B[k][n]=PV[k][n])  [16,256]
  wave_gemm<false>(pvt, actsB, N_PROC, D_PV, gF);
  __syncthreads();
  for (int i = threadIdx.x; i < 16 * D_PV; i += BLOCK)
    gB[i] = __float2bfloat16(gF[i]);
  __syncthreads();

  // Stage 3: A3 = gelu(G @ OW^T)  [16,2048]
  wave_gemm<true>(ow, gB, D_PV, N_OUT, actsF);
  __syncthreads();
  // top-256 threshold mask -> W3 (bf16)
  select_mask(actsF, actsB, N_OUT, K_OUT, hist, rowsel, 0);
  __syncthreads();

  // Stage 3b: OUT = W3 @ OP  (OPT is [512][2048] so B[k][n]=OP[k][n])
  // write f32 result straight to global.
  wave_gemm<false>(opt, actsB, N_OUT, D_MODEL,
                   out + (size_t)tokenBase * D_MODEL);
}

extern "C" void kernel_launch(void* const* d_in, const int* in_sizes, int n_in,
                              void* d_out, int out_size, void* d_ws, size_t ws_size,
                              hipStream_t stream) {
  (void)in_sizes; (void)n_in; (void)out_size; (void)ws_size;
  const float* x   = (const float*)d_in[0];  // [2,1024,512]
  const float* ipf = (const float*)d_in[1];  // [2048,512]
  const float* pwf = (const float*)d_in[2];  // [1024,2048]
  const float* pvf = (const float*)d_in[3];  // [1024,256]
  const float* owf = (const float*)d_in[4];  // [2048,256]
  const float* opf = (const float*)d_in[5];  // [2048,512]
  float* out = (float*)d_out;
  __hip_bfloat16* ws = (__hip_bfloat16*)d_ws;  // needs >= ~9.5 MB

  // opt in to >64KB dynamic LDS (deterministic, capture-safe, no stream use)
  (void)hipFuncSetAttribute((const void*)ffn3_main_kernel,
                            hipFuncAttributeMaxDynamicSharedMemorySize,
                            (int)SM_BYTES);

  // weight precompute: f32 -> bf16 (+transpose for PV, OP)
  {
    int n1 = N_IN * D_MODEL;
    ffn3_cvt_direct<<<(n1 + 255) / 256, 256, 0, stream>>>(ipf, ws + WS_IP, n1);
    int n2 = N_PROC * N_IN;
    ffn3_cvt_direct<<<(n2 + 255) / 256, 256, 0, stream>>>(pwf, ws + WS_PW, n2);
    int n3 = N_OUT * D_PV;
    ffn3_cvt_direct<<<(n3 + 255) / 256, 256, 0, stream>>>(owf, ws + WS_OW, n3);
    int n4 = N_PROC * D_PV;
    ffn3_cvt_transpose<<<(n4 + 255) / 256, 256, 0, stream>>>(pvf, ws + WS_PVT,
                                                             N_PROC, D_PV);
    int n5 = N_OUT * D_MODEL;
    ffn3_cvt_transpose<<<(n5 + 255) / 256, 256, 0, stream>>>(opf, ws + WS_OPT,
                                                             N_OUT, D_MODEL);
  }

  // fused 3-stage FFN: 128 tiles x 16 tokens
  ffn3_main_kernel<<<TOKENS / 16, BLOCK, SM_BYTES, stream>>>(x, ws, out);
}
